// Attention_25795573579904
// MI455X (gfx1250) — matrix operand order repaired and checked by reading, not compiled, not run on hardware
//
#include <hip/hip_runtime.h>
#include <hip/hip_bf16.h>

// ---------------------------------------------------------------------------
// Types for CDNA5 WMMA (wave32): V_WMMA_F32_16X16X32_BF16
// ---------------------------------------------------------------------------
typedef __attribute__((ext_vector_type(16))) __bf16          v16bf;
typedef __attribute__((ext_vector_type(8)))  float           v8f;
typedef __attribute__((ext_vector_type(4)))  unsigned short  v4u;
typedef __attribute__((ext_vector_type(16))) unsigned short  v16u;

#define HEADS 4
#define DH 32
#define NB 2
#define CCH 256
#define NTOK 4096
#define HID 128
#define QK_SCALE 0.17677669529663687f

__device__ __forceinline__ unsigned short f2bf(float f) {
    unsigned u = __builtin_bit_cast(unsigned, f);
    u += 0x7fffu + ((u >> 16) & 1u);          // round-to-nearest-even
    return (unsigned short)(u >> 16);
}

__device__ __forceinline__ v16u combine4(v4u a, v4u b, v4u c, v4u d) {
    v16u r;
#pragma unroll
    for (int i = 0; i < 4; ++i) { r[i] = a[i]; r[4 + i] = b[i]; r[8 + i] = c[i]; r[12 + i] = d[i]; }
    return r;
}

// A operand (16x32 bf16), source row-major [16][stride]:
// lanes 0-15: row m, K {0..7,16..23}; lanes 16-31: row m, K {8..15,24..31}
__device__ __forceinline__ v16bf load_mat_a(const unsigned short* base, int stride, int lane) {
    int m = lane & 15, h8 = (lane >> 4) << 3;
    const unsigned short* p = base + m * stride + h8;
    v4u a0 = *(const v4u*)(p);
    v4u a1 = *(const v4u*)(p + 4);
    v4u a2 = *(const v4u*)(p + 16);
    v4u a3 = *(const v4u*)(p + 20);
    return __builtin_bit_cast(v16bf, combine4(a0, a1, a2, a3));
}

// B operand (32x16 bf16), source stored transposed row-major [N][stride] (K contiguous):
// lanes 0-15: col n, K {0..7,16..23}; lanes 16-31: col n, K {8..15,24..31}
__device__ __forceinline__ v16bf load_mat_b(const unsigned short* base, int stride, int lane) {
    int n = lane & 15, kb = (lane >> 4) << 3;
    const unsigned short* p = base + n * stride + kb;
    v4u a0 = *(const v4u*)(p);
    v4u a1 = *(const v4u*)(p + 4);
    v4u a2 = *(const v4u*)(p + 16);
    v4u a3 = *(const v4u*)(p + 20);
    return __builtin_bit_cast(v16bf, combine4(a0, a1, a2, a3));
}

__device__ __forceinline__ v8f wmma_bf16(v16bf a, v16bf b, v8f c) {
    return __builtin_amdgcn_wmma_f32_16x16x32_bf16(false, a, false, b, (short)0, c, false, false);
}

__device__ __forceinline__ float redmax16(float v) {
    v = fmaxf(v, __shfl_xor(v, 1, 32));
    v = fmaxf(v, __shfl_xor(v, 2, 32));
    v = fmaxf(v, __shfl_xor(v, 4, 32));
    v = fmaxf(v, __shfl_xor(v, 8, 32));
    return v;
}
__device__ __forceinline__ float redsum16(float v) {
    v += __shfl_xor(v, 1, 32);
    v += __shfl_xor(v, 2, 32);
    v += __shfl_xor(v, 4, 32);
    v += __shfl_xor(v, 8, 32);
    return v;
}

// ---------------------------------------------------------------------------
// CDNA5 async global->LDS copy (ASYNCcnt-tracked, no VGPR round trip).
// LDS destination address taken from the vdst VGPR; GV mode: 64-bit global address in VGPR pair.
// ---------------------------------------------------------------------------
__device__ __forceinline__ void async_ld_b128(void* lds_ptr, const void* gptr) {
    unsigned lds_off = (unsigned)(uintptr_t)lds_ptr;   // low 32 bits = LDS offset
    unsigned long long ga = (unsigned long long)(uintptr_t)gptr;
    asm volatile("global_load_async_to_lds_b128 %0, %1, off"
                 :: "v"(lds_off), "v"(ga)
                 : "memory");
}
__device__ __forceinline__ void wait_async0() {
    asm volatile("s_wait_asynccnt 0x0" ::: "memory");
}
__device__ __forceinline__ void wait_async2() {
    asm volatile("s_wait_asynccnt 0x2" ::: "memory");
}

// ---------------------------------------------------------------------------
// Kernel 1: t = silu(time_emb) @ w_mlp.T + b_mlp      -> t_mlp [2][512]
// ---------------------------------------------------------------------------
__global__ __launch_bounds__(256) void k_film(const float* __restrict__ time_emb,
                                              const float* __restrict__ w_mlp,
                                              const float* __restrict__ b_mlp,
                                              float* __restrict__ t_mlp) {
    __shared__ float se[512];
    int b = blockIdx.x;
    for (int i = threadIdx.x; i < 512; i += 256) {
        float v = time_emb[b * 512 + i];
        se[i] = v / (1.0f + __expf(-v));
    }
    __syncthreads();
    for (int o = threadIdx.x; o < 512; o += 256) {
        const float* wr = w_mlp + o * 512;
        float acc = b_mlp[o];
        for (int k = 0; k < 512; ++k) acc += se[k] * wr[k];
        t_mlp[b * 512 + o] = acc;
    }
}

// ---------------------------------------------------------------------------
// Kernel 2: fold FiLM into QKV weights:
//   wq_bf[b][o][c] = bf16(w_qkv[o][c] * (1 + scale[b][c]))
//   qb[b][o]       = sum_c w_qkv[o][c] * shift[b][c]
// ---------------------------------------------------------------------------
__global__ __launch_bounds__(256) void k_prep(const float* __restrict__ w_qkv,
                                              const float* __restrict__ t_mlp,
                                              unsigned short* __restrict__ wq_bf,
                                              float* __restrict__ qb) {
    __shared__ float red[256];
    int bo = blockIdx.x;             // 0..767
    int b = bo / 384, o = bo % 384;
    int c = threadIdx.x;             // 0..255
    float scale = t_mlp[b * 512 + c];
    float shift = t_mlp[b * 512 + 256 + c];
    float w = w_qkv[o * 256 + c];
    wq_bf[(b * 384 + o) * 256 + c] = f2bf(w * (1.0f + scale));
    red[c] = w * shift;
    __syncthreads();
    for (int s = 128; s > 0; s >>= 1) {
        if (c < s) red[c] += red[c + s];
        __syncthreads();
    }
    if (c == 0) qb[b * 384 + o] = red[0];
}

__global__ __launch_bounds__(256) void k_cvt_wout(const float* __restrict__ w_out,
                                                  unsigned short* __restrict__ wout_bf) {
    int i = blockIdx.x * 256 + threadIdx.x;   // 32768 total
    if (i < 256 * 128) wout_bf[i] = f2bf(w_out[i]);
}

// ---------------------------------------------------------------------------
// Kernel 3: QKV GEMM  qkv[b][o][p] = sum_c wq'[o][c] * x[c][p] + qb[o]
// Writes Q[b][h][n][32] (pre-scaled), K[b][h][n][32], Vt[b][h][32][n] in bf16.
// Block: 64 o x 128 p, 8 waves, WMMA bf16 over K=256 in 8 steps.
// ---------------------------------------------------------------------------
#define WP_STRIDE 36
#define XT_STRIDE 36
__global__ __launch_bounds__(256) void k_qkv(const unsigned short* __restrict__ wq_bf,
                                             const float* __restrict__ x,
                                             const float* __restrict__ qb,
                                             unsigned short* __restrict__ Q,
                                             unsigned short* __restrict__ K,
                                             unsigned short* __restrict__ Vt) {
    __shared__ alignas(16) unsigned short sW[64 * WP_STRIDE];
    __shared__ alignas(16) unsigned short sX[128 * XT_STRIDE];
    int pblk = blockIdx.x * 128;
    int oblk = blockIdx.y * 64;
    int b = blockIdx.z;
    int tid = threadIdx.x, lane = tid & 31, wave = tid >> 5;
    int osub = (wave & 3) * 16;
    int phalf = (wave >> 2) * 64;

    const float* xb = x + (size_t)b * CCH * NTOK;
    const unsigned short* wb = wq_bf + ((size_t)b * 384 + oblk) * 256;

    v8f acc[4];
#pragma unroll
    for (int t = 0; t < 4; ++t)
#pragma unroll
        for (int i = 0; i < 8; ++i) acc[t][i] = 0.0f;

    for (int ks = 0; ks < 8; ++ks) {
        __syncthreads();
        {   // stage W' chunk: 64 rows x 32 c
            int row = tid >> 2, seg = tid & 3;
            const unsigned short* src = wb + row * 256 + ks * 32 + seg * 8;
            v4u x0 = *(const v4u*)src;
            v4u x1 = *(const v4u*)(src + 4);
            unsigned short* dst = sW + row * WP_STRIDE + seg * 8;
            *(v4u*)dst = x0;
            *(v4u*)(dst + 4) = x1;
        }
        {   // stage X transposed: sX[p][c] = bf16(x[ks*32+c][pblk+p])
            for (int i = 0; i < 16; ++i) {
                int idx = tid + i * 256;          // 0..4095
                int c = idx >> 7, p = idx & 127;
                float v = xb[(size_t)(ks * 32 + c) * NTOK + pblk + p];
                sX[p * XT_STRIDE + c] = f2bf(v);
            }
        }
        __syncthreads();
        v16bf a = load_mat_a(sW + osub * WP_STRIDE, WP_STRIDE, lane);
#pragma unroll
        for (int t = 0; t < 4; ++t) {
            v16bf bm = load_mat_b(sX + (phalf + t * 16) * XT_STRIDE, XT_STRIDE, lane);
            acc[t] = wmma_bf16(a, bm, acc[t]);
        }
    }
    // epilogue: add folded bias, route to Q/K/Vt
#pragma unroll
    for (int t = 0; t < 4; ++t) {
        int p = pblk + phalf + t * 16 + (lane & 15);
#pragma unroll
        for (int r = 0; r < 8; ++r) {
            int o = oblk + osub + (lane >> 4) * 8 + r;
            float v = acc[t][r] + qb[b * 384 + o];
            int sec = o >> 7;            // 0=q 1=k 2=v
            int oh = o & 127;
            int h = oh >> 5, d = oh & 31;
            size_t bh = (size_t)(b * HEADS + h);
            if (sec == 0)      Q[(bh * NTOK + p) * DH + d] = f2bf(v * QK_SCALE);
            else if (sec == 1) K[(bh * NTOK + p) * DH + d] = f2bf(v);
            else               Vt[(bh * DH + d) * NTOK + p] = f2bf(v);
        }
    }
}

// ---------------------------------------------------------------------------
// Kernel 4: flash attention. Grid (n/128, b*h). 8 waves; wave owns 16 rows.
// K/Vt chunks double-buffered in LDS via async global->LDS B128 copies.
// ---------------------------------------------------------------------------
#define KST 40   // 80B rows: async B128 dst 16B-aligned; 20*n mod 64 distinct
#define VST 72   // 144B rows: async B128 dst 16B-aligned; 36*n mod 64 distinct
#define PST 68
__global__ __launch_bounds__(256) void k_attn(const unsigned short* __restrict__ Q,
                                              const unsigned short* __restrict__ K,
                                              const unsigned short* __restrict__ Vt,
                                              unsigned short* __restrict__ Ot) {
    __shared__ alignas(16) unsigned short sK[2][64 * KST];
    __shared__ alignas(16) unsigned short sV[2][32 * VST];
    __shared__ alignas(16) unsigned short sP[8 * 16 * PST];

    int bh = blockIdx.y;                  // 0..7
    int b = bh >> 2, h = bh & 3;
    int nblk = blockIdx.x * 128;
    int tid = threadIdx.x, lane = tid & 31, wave = tid >> 5;

    const unsigned short* Qb = Q + (size_t)bh * NTOK * DH;
    const unsigned short* Kb = K + (size_t)bh * NTOK * DH;
    const unsigned short* Vb = Vt + (size_t)bh * DH * NTOK;

    // per-thread async staging addresses (one B128 per thread per matrix)
    int krow = tid >> 2, kseg = tid & 3;          // K: 64 rows x 4 x 16B
    int vrow = tid >> 3, vseg = tid & 7;          // V: 32 rows x 8 x 16B
    unsigned short* kdst0 = &sK[0][krow * KST + kseg * 8];
    unsigned short* kdst1 = &sK[1][krow * KST + kseg * 8];
    unsigned short* vdst0 = &sV[0][vrow * VST + vseg * 8];
    unsigned short* vdst1 = &sV[1][vrow * VST + vseg * 8];
    const unsigned short* ksrc = Kb + (size_t)krow * DH + kseg * 8;
    const unsigned short* vsrc = Vb + (size_t)vrow * NTOK + vseg * 8;

    // wave-private Q tile (A layout), loaded once from global
    v16bf qa = load_mat_a(Qb + (size_t)(nblk + wave * 16) * DH, DH, lane);

    v8f o0, o1;
    float m_r[8], l_r[8];
#pragma unroll
    for (int r = 0; r < 8; ++r) { o0[r] = 0.0f; o1[r] = 0.0f; m_r[r] = -1e30f; l_r[r] = 0.0f; }

    unsigned short* pbuf = sP + wave * 16 * PST;
    int rowoff = (lane >> 4) * 8;
    int colb = lane & 15;

    // prologue: issue chunk 0 into buffer 0
    async_ld_b128(kdst0, ksrc);
    async_ld_b128(vdst0, vsrc);

    const int NIT = NTOK / 64;
    for (int it = 0; it < NIT; ++it) {
        int buf = it & 1;
        // issue next chunk into the other buffer (its last readers finished at
        // the barrier that ended the previous iteration)
        if (it + 1 < NIT) {
            size_t joff = (size_t)(it + 1) * 64;
            unsigned short* kd = (buf ? kdst0 : kdst1);
            unsigned short* vd = (buf ? vdst0 : vdst1);
            async_ld_b128(kd, ksrc + joff * DH);   // K rows advance by 64
            async_ld_b128(vd, vsrc + joff);        // V cols advance by 64
            wait_async2();                         // oldest 2 (chunk it) landed
        } else {
            wait_async0();
        }
        __syncthreads();

        const unsigned short* sKb = sK[buf];
        const unsigned short* sVb = sV[buf];

        // S = Q K^T : four 16x16 tiles, full d=32 reduction per WMMA
        v8f s[4];
#pragma unroll
        for (int t = 0; t < 4; ++t) {
            v16bf kb16 = load_mat_b(sKb + t * 16 * KST, KST, lane);
            v8f z;
#pragma unroll
            for (int i = 0; i < 8; ++i) z[i] = 0.0f;
            s[t] = wmma_bf16(qa, kb16, z);
        }

        // online softmax: row max across 4 tiles + 16 lanes
#pragma unroll
        for (int r = 0; r < 8; ++r) {
            float v = fmaxf(fmaxf(s[0][r], s[1][r]), fmaxf(s[2][r], s[3][r]));
            v = redmax16(v);
            float mn = fmaxf(m_r[r], v);
            float corr = __expf(m_r[r] - mn);
            m_r[r] = mn;
            o0[r] *= corr;
            o1[r] *= corr;
            l_r[r] *= corr;
        }

        // P = exp(S - m): stage to LDS (bf16) + row sums
#pragma unroll
        for (int r = 0; r < 8; ++r) {
            float acc = 0.0f;
#pragma unroll
            for (int t = 0; t < 4; ++t) {
                float pv = __expf(s[t][r] - m_r[r]);
                pbuf[(r + rowoff) * PST + t * 16 + colb] = f2bf(pv);
                acc += pv;
            }
            l_r[r] += redsum16(acc);
        }

        // O += P V : P (16x64) x V (64x32) via 4 WMMAs
#pragma unroll
        for (int cc = 0; cc < 2; ++cc) {
            v16bf pa = load_mat_a(pbuf + cc * 32, PST, lane);
            v16bf bv0 = load_mat_b(sVb + cc * 32, VST, lane);
            o0 = wmma_bf16(pa, bv0, o0);
            v16bf bv1 = load_mat_b(sVb + 16 * VST + cc * 32, VST, lane);
            o1 = wmma_bf16(pa, bv1, o1);
        }
        __syncthreads();   // all waves done reading buf before it is refilled
    }

    // finalize: divide by l, write Ot[b][n][128] (bf16, [N][K] layout for out-proj B)
    unsigned short* obase = Ot + (size_t)b * NTOK * HID;
#pragma unroll
    for (int r = 0; r < 8; ++r) {
        float inv = 1.0f / l_r[r];
        int n = nblk + wave * 16 + rowoff + r;
        obase[(size_t)n * HID + h * DH + colb]      = f2bf(o0[r] * inv);
        obase[(size_t)n * HID + h * DH + 16 + colb] = f2bf(o1[r] * inv);
    }
}

// ---------------------------------------------------------------------------
// Kernel 5: out = w_out @ O + b_out   (256 x 128 x 4096 per batch), WMMA bf16,
// A and B read directly from global (bf16, cache-resident).
// ---------------------------------------------------------------------------
__global__ __launch_bounds__(256) void k_oproj(const unsigned short* __restrict__ wout_bf,
                                               const unsigned short* __restrict__ Ot,
                                               const float* __restrict__ b_out,
                                               float* __restrict__ y) {
    int pblk = blockIdx.x * 128;
    int oblk = blockIdx.y * 64;
    int b = blockIdx.z;
    int tid = threadIdx.x, lane = tid & 31, wave = tid >> 5;
    int osub = (wave & 3) * 16;
    int phalf = (wave >> 2) * 64;

    const unsigned short* ob = Ot + (size_t)b * NTOK * HID;

    v8f acc[4];
#pragma unroll
    for (int t = 0; t < 4; ++t)
#pragma unroll
        for (int i = 0; i < 8; ++i) acc[t][i] = 0.0f;

#pragma unroll
    for (int ks = 0; ks < 4; ++ks) {
        v16bf a = load_mat_a(wout_bf + (size_t)(oblk + osub) * HID + ks * 32, HID, lane);
#pragma unroll
        for (int t = 0; t < 4; ++t) {
            v16bf bm = load_mat_b(ob + (size_t)(pblk + phalf + t * 16) * HID + ks * 32, HID, lane);
            acc[t] = wmma_bf16(a, bm, acc[t]);
        }
    }
#pragma unroll
    for (int t = 0; t < 4; ++t) {
        int p = pblk + phalf + t * 16 + (lane & 15);
#pragma unroll
        for (int r = 0; r < 8; ++r) {
            int o = oblk + osub + (lane >> 4) * 8 + r;
            y[((size_t)b * CCH + o) * NTOK + p] = acc[t][r] + b_out[o];
        }
    }
}

// ---------------------------------------------------------------------------
extern "C" void kernel_launch(void* const* d_in, const int* in_sizes, int n_in,
                              void* d_out, int out_size, void* d_ws, size_t ws_size,
                              hipStream_t stream) {
    const float* x        = (const float*)d_in[0];
    const float* time_emb = (const float*)d_in[1];
    const float* w_mlp    = (const float*)d_in[2];
    const float* b_mlp    = (const float*)d_in[3];
    const float* w_qkv    = (const float*)d_in[4];
    const float* w_out    = (const float*)d_in[5];
    const float* b_out    = (const float*)d_in[6];
    float* out = (float*)d_out;

    char* p = (char*)d_ws;
    auto alloc = [&](size_t bytes) -> void* {
        void* r = (void*)p;
        p += (bytes + 255) & ~(size_t)255;
        return r;
    };
    float*          t_mlp   = (float*)alloc((size_t)NB * 512 * 4);
    float*          qb      = (float*)alloc((size_t)NB * 384 * 4);
    unsigned short* wq_bf   = (unsigned short*)alloc((size_t)NB * 384 * 256 * 2);
    unsigned short* wout_bf = (unsigned short*)alloc((size_t)256 * 128 * 2);
    unsigned short* Q       = (unsigned short*)alloc((size_t)NB * HEADS * NTOK * DH * 2);
    unsigned short* K       = (unsigned short*)alloc((size_t)NB * HEADS * NTOK * DH * 2);
    unsigned short* Vt      = (unsigned short*)alloc((size_t)NB * HEADS * DH * NTOK * 2);
    unsigned short* Ot      = (unsigned short*)alloc((size_t)NB * NTOK * HID * 2);

    k_film<<<dim3(NB), dim3(256), 0, stream>>>(time_emb, w_mlp, b_mlp, t_mlp);
    k_prep<<<dim3(NB * 384), dim3(256), 0, stream>>>(w_qkv, t_mlp, wq_bf, qb);
    k_cvt_wout<<<dim3(128), dim3(256), 0, stream>>>(w_out, wout_bf);
    k_qkv<<<dim3(NTOK / 128, 384 / 64, NB), dim3(256), 0, stream>>>(wq_bf, x, qb, Q, K, Vt);
    k_attn<<<dim3(NTOK / 128, NB * HEADS), dim3(256), 0, stream>>>(Q, K, Vt, Ot);
    k_oproj<<<dim3(NTOK / 128, CCH / 64, NB), dim3(256), 0, stream>>>(wout_bf, Ot, b_out, out);
}
